// VQEmbedding_52793738003227
// MI455X (gfx1250) — compile-verified, matrix operand order checked by reading, not run on hardware
//
#include <hip/hip_runtime.h>
#include <hip/hip_bf16.h>

// CDNA5 WMMA operand vectors
typedef __attribute__((ext_vector_type(16))) __bf16 v16bf;
typedef __attribute__((ext_vector_type(8)))  float  v8f;

#define EMBED_NUM 4096
#define EMBED_DIM 64
#define N_INPUTS  65536
#define N_TILES   (EMBED_NUM / 16)   // 256 column tiles
#define M_TILES   (N_INPUTS / 16)    // 4096 row tiles
#define CHUNK     8                  // tiles staged per round
#define N_ROUNDS  (N_TILES / CHUNK)  // 32
#define SCORE_BIAS 8.0f              // makes packed scores positive (see vq_main)

// ---- workspace layout (bytes) ----
// ebf    : bf16[N_TILES * 32 lanes * 32 elems]  (B operand, pre-swizzled, *-2) = 524288
// esq    : float[4096]  (= ||e||^2 + SCORE_BIAS)  @ 524288
// counts : float[4096]            @ 540672   (atomic)
// vsum   : float[4096*64]         @ 557056   (atomic)
#define WS_ESQ    524288
#define WS_CNT    540672
#define WS_VSUM   557056
#define WS_ZERO_FLOATS (4096 + 4096*64)   // counts + vsum contiguous

// ---------------- CDNA5 async global->LDS helpers ----------------
__device__ __forceinline__ void async_ld_b128(unsigned lds_off, const void* g) {
    asm volatile("global_load_async_to_lds_b128 %0, %1, off"
                 :: "v"(lds_off), "v"((unsigned long long)(uintptr_t)g)
                 : "memory");
}
__device__ __forceinline__ void wait_async0() {
    asm volatile("s_wait_asynccnt 0" ::: "memory");
}

// ---------------- zero the atomic scratch ----------------
__global__ void vq_zero(char* ws) {
    int i = blockIdx.x * blockDim.x + threadIdx.x;
    float* p = (float*)(ws + WS_CNT);
    if (i < WS_ZERO_FLOATS) p[i] = 0.0f;
}

// ---------------- repack codebook into WMMA B layout ----------------
// For V_WMMA_F32_16X16X32_BF16, B is 32(K)x16(N). Per lane (N = lane&15):
//   lanes 0-15 : K = {0..7, 16..23};  lanes 16-31 : K = {8..15, 24..31}
// Per column tile t: 32 lanes x 32 bf16 (two v16bf: K 0-31 and K 32-63),
// scaled by -2 so the WMMA accumulator directly yields -2*x.e + ||e||^2 + bias.
__global__ void vq_prep(const float* __restrict__ emb, char* ws) {
    __bf16* ebf = (__bf16*)ws;
    float*  esq = (float*)(ws + WS_ESQ);

    int wave = (blockIdx.x * blockDim.x + threadIdx.x) >> 5;   // tile id 0..255
    int lane = threadIdx.x & 31;
    if (wave >= N_TILES) return;

    int col = lane & 15;
    int n   = wave * 16 + col;
    int kb  = (lane < 16) ? 0 : 8;
    const float* er = emb + (size_t)n * EMBED_DIM;

    const int off[8] = {kb, kb + 4, kb + 16, kb + 20,
                        kb + 32, kb + 36, kb + 48, kb + 52};
    float vals[32];
    float ps = 0.0f;
#pragma unroll
    for (int i = 0; i < 8; ++i) {
        float4 q = *(const float4*)(er + off[i]);
        vals[4*i+0] = q.x; vals[4*i+1] = q.y;
        vals[4*i+2] = q.z; vals[4*i+3] = q.w;
        ps += q.x*q.x + q.y*q.y + q.z*q.z + q.w*q.w;
    }
    v16bf b0, b1;
#pragma unroll
    for (int i = 0; i < 16; ++i) {
        b0[i] = (__bf16)(-2.0f * vals[i]);
        b1[i] = (__bf16)(-2.0f * vals[16 + i]);
    }
    __bf16* dst = ebf + (size_t)wave * 1024 + (size_t)lane * 32;
    *(v16bf*)(dst)      = b0;
    *(v16bf*)(dst + 16) = b1;

    // lane and lane^16 together cover all 64 K values of row n
    ps += __shfl_xor(ps, 16, 32);
    if (lane < 16) esq[n] = ps + SCORE_BIAS;
}

// ---------------- fused distance GEMM + argmin + gather + segment-sum ----------------
// Each wave owns 32 input rows (two 16-row M-tiles) -> 4 WMMAs per staged B tile.
// Argmin uses the packed-u32 trick: scores are positive (biased), so IEEE bit
// order == value order; pack index into the low 12 mantissa bits and v_min_u32.
__global__ void __launch_bounds__(256)
vq_main(const float* __restrict__ x, const float* __restrict__ emb,
        char* ws, float* __restrict__ out) {
    const char* ebfG   = (const char*)ws;                 // 524288 B, tile stride 2048
    float*      counts = (float*)(ws + WS_CNT);
    float*      vsum   = (float*)(ws + WS_VSUM);
    float* out_code  = out;                       // [65536]
    float* out_quant = out + N_INPUTS;            // [65536*64]

    // LDS: ||e||^2 table + double-buffered B chunks  (16KB + 2*16KB = 48KB)
    __shared__ __align__(16) float  esqs[EMBED_NUM];
    __shared__ __align__(16) __bf16 bufB[2][CHUNK * 1024];

    int tid  = threadIdx.x;          // 0..255
    int lane = tid & 31;
    int wav  = tid >> 5;             // 0..7
    int m0   = (blockIdx.x * 8 + wav) * 32;   // 32 rows per wave
    int col  = lane & 15;
    int kb   = (lane < 16) ? 0 : 8;

    // ---- prime: async-stage esq (16KB) and B chunk 0 (16KB) ----
    {
        unsigned esqOff = (unsigned)(uintptr_t)&esqs[0];
        const char* esqG = (const char*)(ws + WS_ESQ);
        unsigned b0Off = (unsigned)(uintptr_t)&bufB[0][0];
#pragma unroll
        for (int j = 0; j < 4; ++j) {
            async_ld_b128(esqOff + tid * 64 + j * 16, esqG + tid * 64 + j * 16);
            async_ld_b128(b0Off  + tid * 64 + j * 16, ebfG + tid * 64 + j * 16);
        }
    }

    // ---- A operands: 2 tiles x 16 rows x 64 K, bf16, per-lane K striping ----
    v16bf a0, a1, a2, a3;
    {
        const int off[8] = {kb, kb + 4, kb + 16, kb + 20,
                            kb + 32, kb + 36, kb + 48, kb + 52};
        const float* xr0 = x + (size_t)(m0 + col) * EMBED_DIM;
        const float* xr1 = x + (size_t)(m0 + 16 + col) * EMBED_DIM;
        float v0[32], v1[32];
#pragma unroll
        for (int i = 0; i < 8; ++i) {
            float4 q = *(const float4*)(xr0 + off[i]);
            v0[4*i+0] = q.x; v0[4*i+1] = q.y; v0[4*i+2] = q.z; v0[4*i+3] = q.w;
            float4 p = *(const float4*)(xr1 + off[i]);
            v1[4*i+0] = p.x; v1[4*i+1] = p.y; v1[4*i+2] = p.z; v1[4*i+3] = p.w;
        }
#pragma unroll
        for (int i = 0; i < 16; ++i) {
            a0[i] = (__bf16)v0[i];  a1[i] = (__bf16)v0[16 + i];
            a2[i] = (__bf16)v1[i];  a3[i] = (__bf16)v1[16 + i];
        }
    }

    unsigned best0[8], best1[8];
#pragma unroll
    for (int r = 0; r < 8; ++r) { best0[r] = 0xFFFFFFFFu; best1[r] = 0xFFFFFFFFu; }

    wait_async0();
    __syncthreads();

    // ---- sweep all 4096 codes; double-buffered async prefetch of B chunks ----
    for (int rnd = 0; rnd < N_ROUNDS; ++rnd) {
        if (rnd + 1 < N_ROUNDS) {   // prefetch next chunk into other buffer
            unsigned dstOff = (unsigned)(uintptr_t)&bufB[(rnd + 1) & 1][0];
            const char* g = ebfG + (size_t)(rnd + 1) * (CHUNK * 2048);
#pragma unroll
            for (int j = 0; j < 4; ++j)
                async_ld_b128(dstOff + tid * 64 + j * 16, g + tid * 64 + j * 16);
        }
        const __bf16* bb = &bufB[rnd & 1][0];
#pragma unroll
        for (int tt = 0; tt < CHUNK; ++tt) {
            int t = rnd * CHUNK + tt;
            const __bf16* bp = bb + tt * 1024 + lane * 32;
            v16bf b0 = *(const v16bf*)(bp);
            v16bf b1 = *(const v16bf*)(bp + 16);
            float ev = esqs[t * 16 + col];           // ||e||^2 + bias
            v8f c0 = {ev, ev, ev, ev, ev, ev, ev, ev};
            v8f c1 = {ev, ev, ev, ev, ev, ev, ev, ev};
            c0 = __builtin_amdgcn_wmma_f32_16x16x32_bf16(false, a0, false, b0,
                                                         (short)0, c0, false, false);
            c0 = __builtin_amdgcn_wmma_f32_16x16x32_bf16(false, a1, false, b1,
                                                         (short)0, c0, false, false);
            c1 = __builtin_amdgcn_wmma_f32_16x16x32_bf16(false, a2, false, b0,
                                                         (short)0, c1, false, false);
            c1 = __builtin_amdgcn_wmma_f32_16x16x32_bf16(false, a3, false, b1,
                                                         (short)0, c1, false, false);
            unsigned n = (unsigned)(t * 16 + col);
#pragma unroll
            for (int r = 0; r < 8; ++r) {
                unsigned p0 = (__float_as_uint(c0[r]) & 0xFFFFF000u) | n;
                unsigned p1 = (__float_as_uint(c1[r]) & 0xFFFFF000u) | n;
                best0[r] = min(best0[r], p0);
                best1[r] = min(best1[r], p1);
            }
        }
        wait_async0();
        __syncthreads();
    }

    // ---- argmin reduction across the 16 lanes holding each row (pure u32 min) ----
#pragma unroll
    for (int offm = 8; offm > 0; offm >>= 1) {
#pragma unroll
        for (int r = 0; r < 8; ++r) {
            best0[r] = min(best0[r], (unsigned)__shfl_xor((int)best0[r], offm, 32));
            best1[r] = min(best1[r], (unsigned)__shfl_xor((int)best1[r], offm, 32));
        }
    }
    // group 0 (lanes 0-15): rows m0+0..7 (tile0) / m0+16..23 (tile1)
    // group 1 (lanes 16-31): rows m0+8..15 / m0+24..31

    if (col < 8) {   // lanes 0-7 and 16-23: two rows each
        int rbase = m0 + col + ((lane < 16) ? 0 : 8);
        int code0 = (int)(best0[col] & 0xFFFu);
        int code1 = (int)(best1[col] & 0xFFFu);
        out_code[rbase]      = (float)code0;
        out_code[rbase + 16] = (float)code1;
        atomicAdd(&counts[code0], 1.0f);   // segment_sum of ones
        atomicAdd(&counts[code1], 1.0f);
    }

    // ---- quant gather + clustered_vectors scatter-add (2 floats per lane) ----
#pragma unroll
    for (int r = 0; r < 32; ++r) {
        int rr  = r & 15;
        int src = (rr < 8) ? 0 : 16;
        unsigned pk = (r < 16) ? (unsigned)__shfl((int)best0[r & 7], src, 32)
                               : (unsigned)__shfl((int)best1[r & 7], src, 32);
        int code = (int)(pk & 0xFFFu);
        size_t eo = (size_t)code * EMBED_DIM + lane * 2;
        size_t ro = (size_t)(m0 + r) * EMBED_DIM + lane * 2;
        float2 e2 = *(const float2*)(emb + eo);
        *(float2*)(out_quant + ro) = e2;
        float2 xv = *(const float2*)(x + ro);
        atomicAdd(&vsum[eo + 0], xv.x);
        atomicAdd(&vsum[eo + 1], xv.y);
    }
}

// ---------------- EMA updates + new embeddings ----------------
__global__ void vq_fin(const float* __restrict__ cs, const float* __restrict__ cv,
                       char* ws, float* __restrict__ out) {
    const float* counts = (const float*)(ws + WS_CNT);
    const float* vsum   = (const float*)(ws + WS_VSUM);
    float* out_emb = out + N_INPUTS + (size_t)N_INPUTS * EMBED_DIM; // [4096*64]
    float* out_cs  = out_emb + EMBED_NUM * EMBED_DIM;               // [4096]
    float* out_cv  = out_cs + EMBED_NUM;                            // [4096*64]

    int i = blockIdx.x * blockDim.x + threadIdx.x;   // 0..262143
    if (i >= EMBED_NUM * EMBED_DIM) return;
    int k = i >> 6, d = i & 63;
    float ncs = 0.9f * cs[k] + 0.1f * counts[k];
    float ncv = 0.9f * cv[i] + 0.1f * vsum[i];
    out_cv[i]  = ncv;
    out_emb[i] = ncv / ncs;
    if (d == 0) out_cs[k] = ncs;
}

extern "C" void kernel_launch(void* const* d_in, const int* in_sizes, int n_in,
                              void* d_out, int out_size, void* d_ws, size_t ws_size,
                              hipStream_t stream) {
    const float* x   = (const float*)d_in[0];   // inputs [65536,64]
    const float* emb = (const float*)d_in[1];   // embeddings [4096,64]
    const float* cs  = (const float*)d_in[2];   // cluster_size [4096]
    const float* cv  = (const float*)d_in[3];   // cluster_vectors [4096,64]
    float* out = (float*)d_out;
    char*  ws  = (char*)d_ws;

    vq_zero<<<(WS_ZERO_FLOATS + 255) / 256, 256, 0, stream>>>(ws);
    vq_prep<<<N_TILES / 8, 256, 0, stream>>>(emb, ws);            // 256 waves
    vq_main<<<M_TILES / 16, 256, 0, stream>>>(x, emb, ws, out);   // 2048 waves, 32 rows each
    vq_fin<<<(EMBED_NUM * EMBED_DIM + 255) / 256, 256, 0, stream>>>(cs, cv, ws, out);
}
